// DWConv3x3_60335700574210
// MI455X (gfx1250) — matrix operand order repaired and checked by reading, not compiled, not run on hardware
//
#include <hip/hip_runtime.h>

// ---------------- problem constants (from reference) ----------------
#define T_STEPS 4
#define BATCH   16
#define CH      384
#define HH      32
#define WW      32
#define N_IMG   (T_STEPS * BATCH * CH)        // 24576 images of 32x32
#define IMG_PIX (HH * WW)                     // 1024
#define ROWS    (BATCH * CH * HH)             // 196608 LIF rows (one per wave)
#define INV_TAU 0.5f
#define VTH     1.0f
#define BN_EPS  1e-5f

typedef __attribute__((ext_vector_type(16))) _Float16 v16h;
typedef __attribute__((ext_vector_type(8)))  float    v8f;

// ---------------- kernel 0: zero the per-channel accumulators ----------------
__global__ void zero_stats_kernel(float* __restrict__ p, int n) {
    int i = blockIdx.x * blockDim.x + threadIdx.x;
    if (i < n) p[i] = 0.0f;
}

// ---------------- kernel 1: LIF membrane scan + bit-pack spikes ----------------
// One wave (32 lanes, lane = w) handles one (b,c,h) row across all T steps.
// v <- v + (x - v)/tau ; spike = v >= vth ; hard reset. Ballot packs 32 spikes
// into one u32 per row per timestep: spike tensor = 3.1 MB, L2-resident.
__global__ __launch_bounds__(256) void lif_pack_kernel(
        const float* __restrict__ x, unsigned* __restrict__ packed) {
    const int lane = threadIdx.x & 31;
    const int wid  = threadIdx.x >> 5;
    const long long row = (long long)blockIdx.x * 8 + wid;   // (b*CH + c)*HH + h
    const size_t tstride = (size_t)BATCH * CH * HH * WW;      // elements per timestep
    const size_t base    = (size_t)row * WW + lane;

    float v = 0.0f;
#pragma unroll
    for (int t = 0; t < T_STEPS; ++t) {
        float xt = x[(size_t)t * tstride + base];
        v = v + (xt - v) * INV_TAU;
        bool spk = (v >= VTH);
        unsigned m = __builtin_amdgcn_ballot_w32(spk);        // bit w = spike at col w
        v = spk ? 0.0f : v;                                   // hard reset
        if (lane == 0)
            packed[(size_t)t * (BATCH * CH * HH) + row] = m;
    }
}

// ---------------- WMMA depthwise 3x3 via Toeplitz B ----------------
// Block = 128 threads = 4 waves; one block per 32x32 image (one (t*B+b, c)).
// Wave w handles output tile (h0,w0) in {0,16}^2:
//   Y(16x16) = sum_{kh=0..2} A_kh(16x32,f16) x B_kh(32x16,f16)
//   A_kh[m][k] = P[h0+m+kh][w0+k]   (P = zero-padded 34x34 spike image, k<=17)
//   B_kh[k][n] = wgt[kh][k-n] for 0<=k-n<=2 else 0  (banded Toeplitz)
// STATS=true : reduce sum / sumsq per channel (for training-mode BN stats).
// STATS=false: fused BN affine (y*scale[c] + bias[c]) -> out.
template <bool STATS>
__global__ __launch_bounds__(128) void dwconv_wmma_kernel(
        const unsigned* __restrict__ packed,
        const float*    __restrict__ w9,          // [CH][3][3]
        float*          __restrict__ out,
        float*          __restrict__ sumBuf,
        float*          __restrict__ sqBuf,
        const float*    __restrict__ scaleBuf,
        const float*    __restrict__ biasBuf) {
    __shared__ unsigned srows[HH];
    __shared__ _Float16 P[34 * 36];               // padded image, row stride 36 halves
    __shared__ float    sAcc[2];

    const int img = blockIdx.x;                   // (t*BATCH+b)*CH + c
    const int c   = img % CH;
    const int tid = threadIdx.x;

    if (tid < HH) srows[tid] = packed[(size_t)img * HH + tid];
    if (STATS && tid == 0) { sAcc[0] = 0.0f; sAcc[1] = 0.0f; }
    __syncthreads();

    // unpack bits -> f16 padded image in LDS
    for (int idx = tid; idx < 34 * 36; idx += 128) {
        int r  = idx / 36;
        int cc = idx - r * 36;
        float val = 0.0f;
        if (r >= 1 && r <= 32 && cc >= 1 && cc <= 32)
            val = (float)((srows[r - 1] >> (cc - 1)) & 1u);
        P[idx] = (_Float16)val;
    }
    __syncthreads();

    const int  lane = tid & 31;
    const int  wave = tid >> 5;
    const int  h0   = (wave >> 1) * 16;
    const int  w0   = (wave & 1) * 16;
    const bool lo   = lane < 16;
    const int  n    = lane & 15;

    // ---- build 3 Toeplitz B fragments (32x16 f16; lanes0-15: K=0..7,16..23, lanes16-31: K=8..15,24..31) ----
    const float* wc = w9 + c * 9;
    v16h bf[3];
#pragma unroll
    for (int kh = 0; kh < 3; ++kh) {
        float t0 = wc[kh * 3 + 0], t1 = wc[kh * 3 + 1], t2 = wc[kh * 3 + 2];
        v16h b;
        int kb = lo ? 0 : 8;
#pragma unroll
        for (int e = 0; e < 16; ++e) {            // element e <-> K = kb + e (+8 if e>=8)
            int d = (kb + e + (e >= 8 ? 8 : 0)) - n;
            float wv = (d == 0) ? t0 : (d == 1) ? t1 : (d == 2) ? t2 : 0.0f;
            b[e] = (_Float16)wv;
        }
        bf[kh] = b;
    }

    // ---- 3 chained WMMAs: acc += A_kh x B_kh ----
    v8f acc = {};
#pragma unroll
    for (int kh = 0; kh < 3; ++kh) {
        v16h a;
        int prow  = h0 + n + kh;                  // 0..33 (input row = out_h + kh - 1, +1 pad)
        int base  = prow * 36 + w0;
        int kbase = lo ? 0 : 8;                   // A 16-bit layout: lanes0-15 K=0..7,16..23
#pragma unroll
        for (int vv = 0; vv < 4; ++vv) {          // elems 0..7  <-> K = kbase+2v, +1
            int K0 = kbase + 2 * vv;
            a[2 * vv]     = P[base + K0];
            a[2 * vv + 1] = P[base + K0 + 1];
        }
        if (lo) { a[8] = P[base + 16]; a[9] = P[base + 17]; }   // K=16,17 still in band
        else    { a[8] = (_Float16)0.0f; a[9] = (_Float16)0.0f; } // K=24.. out of band
#pragma unroll
        for (int e = 10; e < 16; ++e) a[e] = (_Float16)0.0f;     // K>=18 -> zero
        acc = __builtin_amdgcn_wmma_f32_16x16x32_f16(
                  false, a, false, bf[kh], (short)0, acc, false, false);
    }

    // ---- D layout: VGPR r -> M = r + (lane<16 ? 0 : 8), N = lane&15 ----
    if (STATS) {
        float ls = 0.0f, lq = 0.0f;
#pragma unroll
        for (int r = 0; r < 8; ++r) { float yv = acc[r]; ls += yv; lq += yv * yv; }
        atomicAdd(&sAcc[0], ls);
        atomicAdd(&sAcc[1], lq);
        __syncthreads();
        if (tid == 0) {
            atomicAdd(&sumBuf[c], sAcc[0]);
            atomicAdd(&sqBuf[c],  sAcc[1]);
        }
    } else {
        float sc = scaleBuf[c], bi = biasBuf[c];
        size_t obase = (size_t)img * IMG_PIX;
#pragma unroll
        for (int r = 0; r < 8; ++r) {
            int h = h0 + r + (lo ? 0 : 8);
            out[obase + (size_t)h * WW + w0 + n] = acc[r] * sc + bi;
        }
    }
}

// ---------------- kernel 3: per-channel BN scale/bias from batch stats ----------------
__global__ void finalize_stats_kernel(
        const float* __restrict__ sumBuf, const float* __restrict__ sqBuf,
        const float* __restrict__ gamma,  const float* __restrict__ beta,
        float* __restrict__ scaleBuf, float* __restrict__ biasBuf) {
    int c = blockIdx.x * blockDim.x + threadIdx.x;
    if (c < CH) {
        const float cnt = (float)(T_STEPS * BATCH * IMG_PIX);   // 65536 samples/channel
        float mean = sumBuf[c] / cnt;
        float var  = sqBuf[c] / cnt - mean * mean;              // biased var (jnp.var)
        float inv  = rsqrtf(var + BN_EPS) * gamma[c];
        scaleBuf[c] = inv;
        biasBuf[c]  = beta[c] - mean * inv;
    }
}

// ---------------- launcher ----------------
extern "C" void kernel_launch(void* const* d_in, const int* in_sizes, int n_in,
                              void* d_out, int out_size, void* d_ws, size_t ws_size,
                              hipStream_t stream) {
    const float* x     = (const float*)d_in[0];
    const float* w9    = (const float*)d_in[1];
    const float* gamma = (const float*)d_in[2];
    const float* beta  = (const float*)d_in[3];
    float* out = (float*)d_out;

    // workspace layout: packed spikes (3.15 MB) + 4x384 float stats
    const size_t PACKED_WORDS = (size_t)T_STEPS * BATCH * CH * HH;   // 786432 u32
    unsigned* packed = (unsigned*)d_ws;
    float* stats = (float*)((char*)d_ws + PACKED_WORDS * sizeof(unsigned));
    float* sumBuf   = stats;
    float* sqBuf    = stats + CH;
    float* scaleBuf = stats + 2 * CH;
    float* biasBuf  = stats + 3 * CH;

    // 0) zero stats (ws is poisoned; must re-zero every call)
    zero_stats_kernel<<<(2 * CH + 255) / 256, 256, 0, stream>>>(sumBuf, 2 * CH);

    // 1) LIF scan + bit-pack spikes (8 waves / block, one row per wave)
    lif_pack_kernel<<<ROWS / 8, 256, 0, stream>>>(x, packed);

    // 2) depthwise 3x3 via WMMA, accumulate per-channel sum / sumsq
    dwconv_wmma_kernel<true><<<N_IMG, 128, 0, stream>>>(
        packed, w9, nullptr, sumBuf, sqBuf, nullptr, nullptr);

    // 3) finalize BN scale/bias
    finalize_stats_kernel<<<(CH + 255) / 256, 256, 0, stream>>>(
        sumBuf, sqBuf, gamma, beta, scaleBuf, biasBuf);

    // 4) depthwise 3x3 via WMMA again (spikes are L2-resident) + fused BN affine
    dwconv_wmma_kernel<false><<<N_IMG, 128, 0, stream>>>(
        packed, w9, out, nullptr, nullptr, scaleBuf, biasBuf);
}
